// EncoderBlock_16655883174620
// MI455X (gfx1250) — matrix operand order repaired and checked by reading, not compiled, not run on hardware
//
#include <hip/hip_runtime.h>

// ---------------------------------------------------------------------------
// EncoderBlock for MI455X (gfx1250): bf16 WMMA GEMMs + flash attention.
// B=2, T=2048, D=1024, H=16, HS=64.  fp32 in/out, bf16 matrix-core internals.
// ---------------------------------------------------------------------------

typedef __attribute__((ext_vector_type(16))) __bf16 v16bf;
typedef __attribute__((ext_vector_type(8)))  float  v8f;
typedef __attribute__((ext_vector_type(4)))  unsigned int u32x4;

union BF { u32x4 u[2]; v16bf v; };

#define D_  1024
#define T_  2048
#define B_  2
#define H_  16
#define HS_ 64

static __device__ __forceinline__ v8f wmma_bf16(const BF& a, const BF& b, v8f c) {
  return __builtin_amdgcn_wmma_f32_16x16x32_bf16(false, a.v, false, b.v,
                                                 (short)0, c, false, false);
}

// ---------------------------------------------------------------------------
// LayerNorm over D=1024, one block (256 threads) per row, bf16 output.
// ---------------------------------------------------------------------------
__global__ __launch_bounds__(256) void ln_bf16_kernel(
    const float* __restrict__ x, const float* __restrict__ g,
    const float* __restrict__ be, __bf16* __restrict__ out)
{
  const int row = blockIdx.x;
  const int tid = threadIdx.x;
  const float4 v = ((const float4*)(x + (long)row * D_))[tid];
  float s = v.x + v.y + v.z + v.w;
  float q = v.x * v.x + v.y * v.y + v.z * v.z + v.w * v.w;
#pragma unroll
  for (int m = 1; m < 32; m <<= 1) {
    s += __shfl_xor(s, m, 32);
    q += __shfl_xor(q, m, 32);
  }
  __shared__ float rS[8], rQ[8];
  if ((tid & 31) == 0) { rS[tid >> 5] = s; rQ[tid >> 5] = q; }
  __syncthreads();
  float S = 0.f, Q = 0.f;
#pragma unroll
  for (int i = 0; i < 8; ++i) { S += rS[i]; Q += rQ[i]; }
  const float mu   = S * (1.0f / D_);
  const float var  = Q * (1.0f / D_) - mu * mu;
  const float rstd = rsqrtf(var + 1e-5f);
  const float4 gv = ((const float4*)g)[tid];
  const float4 bv = ((const float4*)be)[tid];
  __bf16* o = out + (long)row * D_ + tid * 4;
  o[0] = (__bf16)((v.x - mu) * rstd * gv.x + bv.x);
  o[1] = (__bf16)((v.y - mu) * rstd * gv.y + bv.y);
  o[2] = (__bf16)((v.z - mu) * rstd * gv.z + bv.z);
  o[3] = (__bf16)((v.w - mu) * rstd * gv.w + bv.w);
}

// ---------------------------------------------------------------------------
// LDS-tiled transpose + fp32->bf16 convert: in[R][C] -> out[C][R].
// grid (C/32, R/32), block (32,8).
// ---------------------------------------------------------------------------
__global__ void transpose_cvt_kernel(const float* __restrict__ in,
                                     __bf16* __restrict__ out, int R, int C)
{
  __shared__ float tile[32][33];
  const int r0 = blockIdx.y * 32, c0 = blockIdx.x * 32;
  const int tx = threadIdx.x, ty = threadIdx.y;
#pragma unroll
  for (int i = 0; i < 32; i += 8)
    tile[ty + i][tx] = in[(long)(r0 + ty + i) * C + c0 + tx];
  __syncthreads();
#pragma unroll
  for (int i = 0; i < 32; i += 8)
    out[(long)(c0 + ty + i) * R + r0 + tx] = (__bf16)tile[tx][ty + i];
}

// ---------------------------------------------------------------------------
// Q/K/V weight fold: w[H=16][D=1024][HS=64] fp32 -> out[N=1024][K=1024] bf16
// with out[h*64+s][d] = w[h][d][s].  grid (2, 32, 16), block (32,8).
// ---------------------------------------------------------------------------
__global__ void qkv_cvt_kernel(const float* __restrict__ w,
                               __bf16* __restrict__ out)
{
  __shared__ float tile[32][33];
  const int h  = blockIdx.z;
  const int d0 = blockIdx.y * 32, s0 = blockIdx.x * 32;
  const int tx = threadIdx.x, ty = threadIdx.y;
#pragma unroll
  for (int i = 0; i < 32; i += 8)
    tile[ty + i][tx] = w[(long)h * D_ * HS_ + (long)(d0 + ty + i) * HS_ + s0 + tx];
  __syncthreads();
#pragma unroll
  for (int i = 0; i < 32; i += 8)
    out[(long)(h * HS_ + s0 + ty + i) * D_ + d0 + tx] = (__bf16)tile[tx][ty + i];
}

// ---------------------------------------------------------------------------
// Tiled bf16 GEMM: C[M,N] = A[M,K] @ BT[N,K]^T  (fp32 accumulate via WMMA).
// Workgroup tile 128x128, 8 waves (4x2), wave tile 32x64 (2x4 WMMA tiles).
// Double-buffered LDS, padded row stride (40 bf16 = 80B, 16B aligned).
// Epilogue: optional bias[n], resid[m,n], ReLU; fp32 and/or bf16 output.
// ---------------------------------------------------------------------------
#define GLDA 40

__global__ __launch_bounds__(256) void gemm_bf16_kernel(
    const __bf16* __restrict__ A, const __bf16* __restrict__ BT,
    const float* __restrict__ bias, const float* __restrict__ resid,
    float* __restrict__ outf, __bf16* __restrict__ outb,
    int M, int N, int K, int relu)
{
  __shared__ __align__(16) __bf16 As[2][128 * GLDA];
  __shared__ __align__(16) __bf16 Bs[2][128 * GLDA];

  const int tid  = threadIdx.x;
  const int wave = tid >> 5;
  const int lane = tid & 31;
  const int l    = lane & 15;
  const int hf   = lane >> 4;
  const int wm0  = (wave >> 1) * 32;   // wave M origin in block tile
  const int wn0  = (wave & 1) * 64;    // wave N origin in block tile
  const long m0  = (long)blockIdx.y * 128;
  const long n0  = (long)blockIdx.x * 128;

  v8f acc[2][4];
  const v8f vz = {0.f, 0.f, 0.f, 0.f, 0.f, 0.f, 0.f, 0.f};
#pragma unroll
  for (int i = 0; i < 2; ++i)
#pragma unroll
    for (int j = 0; j < 4; ++j) acc[i][j] = vz;

  const int KT = K / 32;

  auto loadStage = [&](int s, int kt) {
#pragma unroll
    for (int j = 0; j < 2; ++j) {
      const int cid = tid + j * 256;          // 512 16B chunks per tile
      const int row = cid >> 2;               // 0..127
      const int c   = cid & 3;                // 0..3 (x8 bf16 = 16B)
      *(u32x4*)(&As[s][row * GLDA + c * 8]) =
          *(const u32x4*)(A + (m0 + row) * (long)K + (long)kt * 32 + c * 8);
      *(u32x4*)(&Bs[s][row * GLDA + c * 8]) =
          *(const u32x4*)(BT + (n0 + row) * (long)K + (long)kt * 32 + c * 8);
    }
  };

  loadStage(0, 0);
  __syncthreads();

  for (int kt = 0; kt < KT; ++kt) {
    const int s = kt & 1;
    if (kt + 1 < KT) loadStage(s ^ 1, kt + 1);

    // A fragments: lane holds row M, interleaved K chunks per lane-half.
    BF af[2];
#pragma unroll
    for (int i = 0; i < 2; ++i) {
      const __bf16* p = &As[s][(wm0 + i * 16 + l) * GLDA];
      af[i].u[0] = *(const u32x4*)(p + hf * 8);
      af[i].u[1] = *(const u32x4*)(p + 16 + hf * 8);
    }
    // B fragments: lane holds column N, interleaved K chunks per lane-half.
    BF bfr[4];
#pragma unroll
    for (int j = 0; j < 4; ++j) {
      const __bf16* p = &Bs[s][(wn0 + j * 16 + l) * GLDA];
      bfr[j].u[0] = *(const u32x4*)(p + hf * 8);
      bfr[j].u[1] = *(const u32x4*)(p + 16 + hf * 8);
    }
#pragma unroll
    for (int i = 0; i < 2; ++i)
#pragma unroll
      for (int j = 0; j < 4; ++j)
        acc[i][j] = wmma_bf16(af[i], bfr[j], acc[i][j]);
    __syncthreads();
  }

  // Epilogue.  C layout: VGPR r -> M = 8*hf + r, lane -> N = l.
#pragma unroll
  for (int i = 0; i < 2; ++i)
#pragma unroll
    for (int j = 0; j < 4; ++j)
#pragma unroll
      for (int r = 0; r < 8; ++r) {
        const long m = m0 + wm0 + i * 16 + hf * 8 + r;
        const long n = n0 + wn0 + j * 16 + l;
        float v = acc[i][j][r];
        if (bias)  v += bias[n];
        if (resid) v += resid[m * N + n];
        if (relu)  v = v > 0.f ? v : 0.f;
        if (outf)  outf[m * N + n] = v;
        if (outb)  outb[m * N + n] = (__bf16)v;
      }
}

// ---------------------------------------------------------------------------
// Flash attention: grid (T/128, H, B), 256 threads (8 waves).
// Each wave owns 16 query rows; iterates 64-key blocks with online softmax.
// q/k/v/o stored as [B*T, D] bf16 with head h at column offset h*64.
// ---------------------------------------------------------------------------
#define AP 88   // padded LDS row stride in bf16 (176B = 11*16B, conflict-free)

__global__ __launch_bounds__(256) void attn_kernel(
    const __bf16* __restrict__ q, const __bf16* __restrict__ k,
    const __bf16* __restrict__ v, const int* __restrict__ mask,
    __bf16* __restrict__ o)
{
  __shared__ __align__(16) __bf16 Qs[128 * AP];
  __shared__ __align__(16) __bf16 Ks[64 * AP];
  __shared__ __align__(16) __bf16 Vt[64 * AP];      // [s][key] (transposed V)
  __shared__ __align__(16) __bf16 Ps[8][16 * AP];   // per-wave P tile

  const int tid  = threadIdx.x;
  const int wave = tid >> 5;
  const int lane = tid & 31;
  const int l    = lane & 15;
  const int hf   = lane >> 4;
  const int h    = blockIdx.y;
  const int b    = blockIdx.z;
  const long m0  = (long)blockIdx.x * 128;

  const __bf16* qbase = q + ((long)b * T_) * D_ + h * HS_;
  const __bf16* kbase = k + ((long)b * T_) * D_ + h * HS_;
  const __bf16* vbase = v + ((long)b * T_) * D_ + h * HS_;

  // Stage Q block: 128 rows x 64 cols bf16 = 1024 x 16B chunks.
#pragma unroll
  for (int j = 0; j < 4; ++j) {
    const int cid = tid + j * 256;
    const int row = cid >> 3, c = cid & 7;
    *(u32x4*)(&Qs[row * AP + c * 8]) =
        *(const u32x4*)(qbase + (m0 + row) * D_ + c * 8);
  }
  __syncthreads();

  // Per-wave Q fragments (2 K-chunks of 32 over HS=64), held for whole loop.
  BF qf[2];
#pragma unroll
  for (int c = 0; c < 2; ++c) {
    const __bf16* p = &Qs[(wave * 16 + l) * AP + c * 32];
    qf[c].u[0] = *(const u32x4*)(p + hf * 8);
    qf[c].u[1] = *(const u32x4*)(p + 16 + hf * 8);
  }

  const v8f vz = {0.f, 0.f, 0.f, 0.f, 0.f, 0.f, 0.f, 0.f};
  v8f ov[4];
#pragma unroll
  for (int st = 0; st < 4; ++st) ov[st] = vz;
  float mx[8], sm[8];
#pragma unroll
  for (int r = 0; r < 8; ++r) { mx[r] = -3.0e38f; sm[r] = 0.f; }
  const float scale = 0.125f;  // 1/sqrt(64)

  for (int kb = 0; kb < T_ / 64; ++kb) {
    __syncthreads();  // protect K/V tiles from previous iteration's readers
    // Stage K tile (row-major) and V tile (transposed into Vt[s][key]).
#pragma unroll
    for (int j = 0; j < 2; ++j) {
      const int cid = tid + j * 256;
      const int row = cid >> 3, c = cid & 7;
      *(u32x4*)(&Ks[row * AP + c * 8]) =
          *(const u32x4*)(kbase + ((long)kb * 64 + row) * D_ + c * 8);
      u32x4 d = *(const u32x4*)(vbase + ((long)kb * 64 + row) * D_ + c * 8);
      const __bf16* e = (const __bf16*)&d;
#pragma unroll
      for (int i = 0; i < 8; ++i) Vt[(c * 8 + i) * AP + row] = e[i];
    }
    __syncthreads();

    // S = Q @ K^T : 16 x 64 per wave (4 N-tiles, 2 K-chunks over HS).
    v8f sv[4];
#pragma unroll
    for (int j = 0; j < 4; ++j) sv[j] = vz;
#pragma unroll
    for (int j = 0; j < 4; ++j)
#pragma unroll
      for (int c = 0; c < 2; ++c) {
        BF bk;
        const __bf16* p = &Ks[(j * 16 + l) * AP + c * 32];
        bk.u[0] = *(const u32x4*)(p + hf * 8);
        bk.u[1] = *(const u32x4*)(p + 16 + hf * 8);
        sv[j] = wmma_bf16(qf[c], bk, sv[j]);
      }

    // Scale + mask + row max (rows live in VGPR index r per lane-half).
    float pv[4][8], rm[8];
#pragma unroll
    for (int r = 0; r < 8; ++r) rm[r] = -3.0e38f;
#pragma unroll
    for (int j = 0; j < 4; ++j)
#pragma unroll
      for (int r = 0; r < 8; ++r) {
        float val = sv[j][r] * scale;
        const long mrow = m0 + wave * 16 + hf * 8 + r;
        const long ncol = (long)kb * 64 + j * 16 + l;
        if (mask[mrow * T_ + ncol] == 0) val = -3.0e38f;
        pv[j][r] = val;
        rm[r] = fmaxf(rm[r], val);
      }
#pragma unroll
    for (int r = 0; r < 8; ++r) {
      float m_ = rm[r];
      m_ = fmaxf(m_, __shfl_xor(m_, 1, 32));
      m_ = fmaxf(m_, __shfl_xor(m_, 2, 32));
      m_ = fmaxf(m_, __shfl_xor(m_, 4, 32));
      m_ = fmaxf(m_, __shfl_xor(m_, 8, 32));
      rm[r] = m_;
    }

    // Online softmax update.
    float corr[8], rs[8];
#pragma unroll
    for (int r = 0; r < 8; ++r) {
      const float nm = fmaxf(mx[r], rm[r]);
      corr[r] = __expf(mx[r] - nm);
      mx[r] = nm;
      sm[r] *= corr[r];
      rs[r] = 0.f;
    }
#pragma unroll
    for (int st = 0; st < 4; ++st)
#pragma unroll
      for (int r = 0; r < 8; ++r) ov[st][r] *= corr[r];

#pragma unroll
    for (int j = 0; j < 4; ++j)
#pragma unroll
      for (int r = 0; r < 8; ++r) {
        const float p = __expf(pv[j][r] - mx[r]);
        rs[r] += p;
        Ps[wave][(hf * 8 + r) * AP + j * 16 + l] = (__bf16)p;
      }
#pragma unroll
    for (int r = 0; r < 8; ++r) {
      float s_ = rs[r];
      s_ += __shfl_xor(s_, 1, 32);
      s_ += __shfl_xor(s_, 2, 32);
      s_ += __shfl_xor(s_, 4, 32);
      s_ += __shfl_xor(s_, 8, 32);
      sm[r] += s_;
    }

    // O += P @ V : P is 16x64 (k-dim = keys), Vt gives the B operand.
    BF pf[2];
#pragma unroll
    for (int c = 0; c < 2; ++c) {
      const __bf16* p = &Ps[wave][l * AP + c * 32];
      pf[c].u[0] = *(const u32x4*)(p + hf * 8);
      pf[c].u[1] = *(const u32x4*)(p + 16 + hf * 8);
    }
#pragma unroll
    for (int st = 0; st < 4; ++st)
#pragma unroll
      for (int c = 0; c < 2; ++c) {
        BF bv;
        const __bf16* p = &Vt[(st * 16 + l) * AP + c * 32];
        bv.u[0] = *(const u32x4*)(p + hf * 8);
        bv.u[1] = *(const u32x4*)(p + 16 + hf * 8);
        ov[st] = wmma_bf16(pf[c], bv, ov[st]);
      }
  }

  // Normalize and write concat-head output (bf16).
  __bf16* obase = o + ((long)b * T_) * D_ + h * HS_;
#pragma unroll
  for (int st = 0; st < 4; ++st)
#pragma unroll
    for (int r = 0; r < 8; ++r) {
      const long m = m0 + wave * 16 + hf * 8 + r;
      obase[m * D_ + st * 16 + l] = (__bf16)(ov[st][r] / sm[r]);
    }
}

// ---------------------------------------------------------------------------
// Host-side orchestration.
// ---------------------------------------------------------------------------
extern "C" void kernel_launch(void* const* d_in, const int* in_sizes, int n_in,
                              void* d_out, int out_size, void* d_ws, size_t ws_size,
                              hipStream_t stream)
{
  (void)in_sizes; (void)n_in; (void)out_size; (void)ws_size;

  const float* x    = (const float*)d_in[0];
  const int*   mask = (const int*)d_in[1];
  const float* Wq   = (const float*)d_in[2];
  const float* Wk   = (const float*)d_in[3];
  const float* Wv   = (const float*)d_in[4];
  const float* Wp   = (const float*)d_in[5];
  const float* bp   = (const float*)d_in[6];
  const float* W1   = (const float*)d_in[7];
  const float* b1   = (const float*)d_in[8];
  const float* W2   = (const float*)d_in[9];
  const float* b2   = (const float*)d_in[10];
  const float* g1   = (const float*)d_in[11];
  const float* be1  = (const float*)d_in[12];
  const float* g2   = (const float*)d_in[13];
  const float* be2  = (const float*)d_in[14];
  float* out = (float*)d_out;

  char* ws = (char*)d_ws;
  size_t off = 0;
  auto alloc = [&](size_t bytes) -> char* {
    char* p = ws + off;
    off += (bytes + 255) & ~(size_t)255;
    return p;
  };

  const size_t MT = (size_t)B_ * T_;                 // 4096 rows
  __bf16* WqT = (__bf16*)alloc((size_t)D_ * D_ * 2);
  __bf16* WkT = (__bf16*)alloc((size_t)D_ * D_ * 2);
  __bf16* WvT = (__bf16*)alloc((size_t)D_ * D_ * 2);
  __bf16* WpT = (__bf16*)alloc((size_t)D_ * D_ * 2);
  __bf16* W1T = (__bf16*)alloc((size_t)4 * D_ * D_ * 2);  // [4096][1024]
  __bf16* W2T = (__bf16*)alloc((size_t)4 * D_ * D_ * 2);  // [1024][4096]
  __bf16* hb  = (__bf16*)alloc(MT * D_ * 2);              // LN output (reused)
  __bf16* qb  = (__bf16*)alloc(MT * D_ * 2);
  __bf16* kb  = (__bf16*)alloc(MT * D_ * 2);
  __bf16* vb  = (__bf16*)alloc(MT * D_ * 2);
  __bf16* ab  = (__bf16*)alloc(MT * D_ * 2);              // attention output
  __bf16* a1  = (__bf16*)alloc(MT * (size_t)(4 * D_) * 2);
  float*  x1  = (float*)alloc(MT * D_ * 4);               // post-attn residual

  const dim3 blkT(32, 8);

  // Weight prep (bf16, transposed [N][K] layouts).
  qkv_cvt_kernel<<<dim3(2, 32, 16), blkT, 0, stream>>>(Wq, WqT);
  qkv_cvt_kernel<<<dim3(2, 32, 16), blkT, 0, stream>>>(Wk, WkT);
  qkv_cvt_kernel<<<dim3(2, 32, 16), blkT, 0, stream>>>(Wv, WvT);
  transpose_cvt_kernel<<<dim3(32, 32),  blkT, 0, stream>>>(Wp, WpT, D_, D_);
  transpose_cvt_kernel<<<dim3(128, 32), blkT, 0, stream>>>(W1, W1T, D_, 4 * D_);
  transpose_cvt_kernel<<<dim3(32, 128), blkT, 0, stream>>>(W2, W2T, 4 * D_, D_);

  // ---- attention sublayer ----
  ln_bf16_kernel<<<MT, 256, 0, stream>>>(x, g1, be1, hb);
  gemm_bf16_kernel<<<dim3(8, 32), 256, 0, stream>>>(hb, WqT, nullptr, nullptr,
      nullptr, qb, (int)MT, D_, D_, 0);
  gemm_bf16_kernel<<<dim3(8, 32), 256, 0, stream>>>(hb, WkT, nullptr, nullptr,
      nullptr, kb, (int)MT, D_, D_, 0);
  gemm_bf16_kernel<<<dim3(8, 32), 256, 0, stream>>>(hb, WvT, nullptr, nullptr,
      nullptr, vb, (int)MT, D_, D_, 0);
  attn_kernel<<<dim3(T_ / 128, H_, B_), 256, 0, stream>>>(qb, kb, vb, mask, ab);
  gemm_bf16_kernel<<<dim3(8, 32), 256, 0, stream>>>(ab, WpT, bp, x,
      x1, nullptr, (int)MT, D_, D_, 0);

  // ---- feed-forward sublayer ----
  ln_bf16_kernel<<<MT, 256, 0, stream>>>(x1, g2, be2, hb);
  gemm_bf16_kernel<<<dim3(32, 32), 256, 0, stream>>>(hb, W1T, b1, nullptr,
      nullptr, a1, (int)MT, 4 * D_, D_, 1);
  gemm_bf16_kernel<<<dim3(8, 32), 256, 0, stream>>>(a1, W2T, b2, x1,
      out, nullptr, (int)MT, D_, 4 * D_, 0);
}
